// MultiHeadAttentionXL_26190710571288
// MI455X (gfx1250) — matrix operand order repaired and checked by reading, not compiled, not run on hardware
//
#include <hip/hip_runtime.h>
#include <hip/hip_bf16.h>

typedef __attribute__((ext_vector_type(16))) __bf16 v16bf;
typedef __attribute__((ext_vector_type(8)))  float  v8f;
typedef __attribute__((ext_vector_type(4)))  unsigned int v4u;

union V16U { v16bf v; v4u q[2]; };

namespace {
enum : int {
  D_IN  = 1024,
  NH    = 16,
  DHD   = 64,
  SEQ   = 1024,
  MEMN  = 1024,
  BATCH = 2,
  TD    = MEMN + SEQ,   // 2048
  HD    = NH * DHD,     // 1024
  N_KV  = 2 * HD,       // 2048
};
constexpr float ATT_SCALE = 0.125f;  // 1/sqrt(64)
}

// ---------------------------------------------------------------- conversions

__global__ __launch_bounds__(256) void xl_cvt_bf16(const float* __restrict__ src,
                                                   __bf16* __restrict__ dst, int n) {
  int i = blockIdx.x * blockDim.x + threadIdx.x;
  if (i < n) dst[i] = (__bf16)src[i];
}

// f32 [rows][cols] -> bf16 [cols][rows]  (transpose-convert for B operands)
__global__ __launch_bounds__(256) void xl_cvtT_bf16(const float* __restrict__ src,
                                                    __bf16* __restrict__ dst,
                                                    int rows, int cols) {
  int i = blockIdx.x * blockDim.x + threadIdx.x;
  if (i >= rows * cols) return;
  int r = i / cols, c = i - r * cols;
  dst[(size_t)c * rows + r] = (__bf16)src[i];
}

// kv f32 [T][B][2*HD] -> k bf16 [B][H][T][DHD] (row-major) and
//                        vT bf16 [B][H][DHD][T] (transposed for WMMA B operand)
__global__ __launch_bounds__(256) void xl_pack_kv(const float* __restrict__ kv,
                                                  __bf16* __restrict__ kb,
                                                  __bf16* __restrict__ vtb) {
  int idx = blockIdx.x * blockDim.x + threadIdx.x;   // over T*B*HD
  if (idx >= TD * BATCH * HD) return;
  int d  = idx & (DHD - 1);
  int h  = (idx >> 6) & (NH - 1);
  int bt = idx >> 10;            // t*BATCH + b
  int b  = bt & (BATCH - 1);
  int t  = bt >> 1;
  size_t src = (size_t)bt * N_KV + h * DHD + d;
  kb [((size_t)(b * NH + h) * TD + t) * DHD + d] = (__bf16)kv[src];
  vtb[((size_t)(b * NH + h) * DHD + d) * TD + t] = (__bf16)kv[src + HD];
}

// q f32 [S][B][HD] + u,v -> qu bf16 [B][H][S][DHD], qv f32 [B][H][S][DHD]
__global__ __launch_bounds__(256) void xl_pack_q(const float* __restrict__ q,
                                                 const float* __restrict__ u,
                                                 const float* __restrict__ v,
                                                 __bf16* __restrict__ qub,
                                                 float* __restrict__ qvb) {
  int idx = blockIdx.x * blockDim.x + threadIdx.x;   // over S*B*HD
  if (idx >= SEQ * BATCH * HD) return;
  int d  = idx & (DHD - 1);
  int h  = (idx >> 6) & (NH - 1);
  int ib = idx >> 10;            // i*BATCH + b
  int b  = ib & (BATCH - 1);
  int i  = ib >> 1;
  float qe = q[idx];
  size_t dst = ((size_t)(b * NH + h) * SEQ + i) * DHD + d;
  qub[dst] = (__bf16)(qe + u[h * DHD + d]);
  qvb[dst] = qe + v[h * DHD + d];
}

// ------------------------------------------------------------- generic GEMM
// C[M][N] f32 = A[M][K] bf16 @ B[K][N] bf16 with B given TRANSPOSED (BT[N][K]).
// One wave computes a 16x64 tile: A loaded once per k-step, 4 WMMAs.
__global__ __launch_bounds__(32) void xl_gemm_bf16(const __bf16* __restrict__ A,
                                                   const __bf16* __restrict__ BT,
                                                   float* __restrict__ C,
                                                   int M, int N, int K) {
  const int lane = threadIdx.x;
  const int hi = lane >> 4, lo = lane & 15;
  const int m0 = blockIdx.y * 16, n0 = blockIdx.x * 64;
  const __bf16* arow = A + (size_t)(m0 + lo) * K;     // A-operand row (m = lo)
  const __bf16* brow0 = BT + (size_t)(n0 +  0 + lo) * K;
  const __bf16* brow1 = BT + (size_t)(n0 + 16 + lo) * K;
  const __bf16* brow2 = BT + (size_t)(n0 + 32 + lo) * K;
  const __bf16* brow3 = BT + (size_t)(n0 + 48 + lo) * K;
  const int kbA = hi * 8;        // unused; A loads use kbB
  const int kbB = hi * 16;       // B per-lane chunk base
  v8f acc0{}, acc1{}, acc2{}, acc3{};
  for (int k0 = 0; k0 < K; k0 += 32) {
    V16U av;
    av.q[0] = *(const v4u*)(arow + k0 + kbB);          // k = kbB..kbB+7
    av.q[1] = *(const v4u*)(arow + k0 + kbB + 8);      // k = kbB+8..kbB+15
    v16bf b0 = *(const v16bf*)(brow0 + k0 + kbB);
    v16bf b1 = *(const v16bf*)(brow1 + k0 + kbB);
    v16bf b2 = *(const v16bf*)(brow2 + k0 + kbB);
    v16bf b3 = *(const v16bf*)(brow3 + k0 + kbB);
    acc0 = __builtin_amdgcn_wmma_f32_16x16x32_bf16(false, av.v, false, b0, (short)0, acc0, false, false);
    acc1 = __builtin_amdgcn_wmma_f32_16x16x32_bf16(false, av.v, false, b1, (short)0, acc1, false, false);
    acc2 = __builtin_amdgcn_wmma_f32_16x16x32_bf16(false, av.v, false, b2, (short)0, acc2, false, false);
    acc3 = __builtin_amdgcn_wmma_f32_16x16x32_bf16(false, av.v, false, b3, (short)0, acc3, false, false);
  }
#pragma unroll
  for (int r = 0; r < 8; ++r) {
    float* crow = C + (size_t)(m0 + r + 8 * hi) * N + n0 + lo;
    crow[ 0] = acc0[r];
    crow[16] = acc1[r];
    crow[32] = acc2[r];
    crow[48] = acc3[r];
  }
}

// --------------------------------------------------- fused flash attention
// grid = (SEQ/16, H, B); 1 wave per block. Streams j in 32-wide blocks:
//   content = WMMA(QU, K^T); pos = exact rel_shift scalar dot; online softmax;
//   P@V via WMMA (V pre-transposed) after an LDS 16x32 transpose of P.
__global__ __launch_bounds__(32) void xl_flash_attn(
    const __bf16* __restrict__ qu,   // [B][H][S][DHD] bf16 (q+u)
    const float*  __restrict__ qv,   // [B][H][S][DHD] f32  (q+v)
    const __bf16* __restrict__ kbh,  // [B][H][T][DHD]
    const __bf16* __restrict__ vtbh, // [B][H][DHD][T]  (transposed V)
    const float*  __restrict__ rmat, // [T][H*DHD] f32 (pos_emb @ W_pos)
    float* __restrict__ ctx)         // [S][B][H*DHD]
{
  const int lane = threadIdx.x;
  const int i0 = blockIdx.x * 16;
  const int h  = blockIdx.y;
  const int b  = blockIdx.z;
  const int bh = b * NH + h;
  const int hi = lane >> 4, lo = lane & 15;
  const int kbA = hi * 8, kbB = hi * 16;

  const __bf16* quB = qu   + (size_t)bh * SEQ * DHD;
  const float*  qvB = qv   + (size_t)bh * SEQ * DHD;
  const __bf16* kB  = kbh  + (size_t)bh * TD * DHD;
  const __bf16* vTB = vtbh + (size_t)bh * DHD * TD;
  const float*  rB  = rmat + h * DHD;           // row stride H*DHD

  // A operand: QU rows i0..i0+15, split into k=[0,32) and k=[32,64)
  V16U qa0, qa1;
  {
    const __bf16* rowp = quB + (size_t)(i0 + lo) * DHD;
    qa0.q[0] = *(const v4u*)(rowp + kbB);
    qa0.q[1] = *(const v4u*)(rowp + kbB + 8);
    qa1.q[0] = *(const v4u*)(rowp + 32 + kbB);
    qa1.q[1] = *(const v4u*)(rowp + 40 + kbB);
  }

  v8f acc0{}, acc1{}, acc2{}, acc3{};
  float mrow[8], lrow[8];
#pragma unroll
  for (int r = 0; r < 8; ++r) { mrow[r] = -3.0e38f; lrow[r] = 0.f; }

  __shared__ __bf16 Plds[16][32];

  for (int j0 = 0; j0 < TD; j0 += 32) {
    // ---- content scores for 16 rows x 32 cols (two 16x16 C tiles)
    v8f s0{}, s1{};
    {
      const __bf16* kr0 = kB + (size_t)(j0 + lo) * DHD;
      v16bf blo = *(const v16bf*)(kr0 + kbB);
      v16bf bhi = *(const v16bf*)(kr0 + 32 + kbB);
      s0 = __builtin_amdgcn_wmma_f32_16x16x32_bf16(false, qa0.v, false, blo, (short)0, s0, false, false);
      s0 = __builtin_amdgcn_wmma_f32_16x16x32_bf16(false, qa1.v, false, bhi, (short)0, s0, false, false);
      const __bf16* kr1 = kB + (size_t)(j0 + 16 + lo) * DHD;
      v16bf clo = *(const v16bf*)(kr1 + kbB);
      v16bf chi = *(const v16bf*)(kr1 + 32 + kbB);
      s1 = __builtin_amdgcn_wmma_f32_16x16x32_bf16(false, qa0.v, false, clo, (short)0, s1, false, false);
      s1 = __builtin_amdgcn_wmma_f32_16x16x32_bf16(false, qa1.v, false, chi, (short)0, s1, false, false);
    }

    // ---- position term with exact Transformer-XL rel_shift (incl. wrap)
#pragma unroll
    for (int c = 0; c < 2; ++c) {
      v8f& sc = c ? s1 : s0;
#pragma unroll
      for (int r = 0; r < 8; ++r) {
        int i = i0 + r + 8 * hi;
        int j = j0 + c * 16 + lo;
        int rel = j - i;
        float p = 0.f;
        if (rel != SEQ + 1) {           // rel == S+1 -> exact zero from shift pad
          int ip, jj;
          if (rel <= SEQ) { ip = i;     jj = rel + (SEQ - 1); }  // main diag band
          else            { ip = i + 1; jj = rel - (SEQ + 2); }  // wrapped band
          const float* qrow = qvB + (size_t)ip * DHD;
          const float* rrow = rB + (size_t)jj * HD;
#pragma unroll
          for (int d = 0; d < DHD; d += 4) {
            float4 a4 = *(const float4*)(qrow + d);
            float4 b4 = *(const float4*)(rrow + d);
            p += a4.x * b4.x + a4.y * b4.y + a4.z * b4.z + a4.w * b4.w;
          }
        }
        sc[r] = (sc[r] + p) * ATT_SCALE;
      }
    }

    // ---- online softmax over these 32 columns
#pragma unroll
    for (int r = 0; r < 8; ++r) {
      float tmax = fmaxf(s0[r], s1[r]);
#pragma unroll
      for (int off = 1; off < 16; off <<= 1)
        tmax = fmaxf(tmax, __shfl_xor(tmax, off, 32));
      float mnew = fmaxf(mrow[r], tmax);
      float corr = __expf(mrow[r] - mnew);
      float p0 = __expf(s0[r] - mnew);
      float p1 = __expf(s1[r] - mnew);
      float rs = p0 + p1;
#pragma unroll
      for (int off = 1; off < 16; off <<= 1)
        rs += __shfl_xor(rs, off, 32);
      lrow[r] = lrow[r] * corr + rs;
      mrow[r] = mnew;
      acc0[r] *= corr; acc1[r] *= corr; acc2[r] *= corr; acc3[r] *= corr;
      Plds[r + 8 * hi][lo]      = (__bf16)p0;
      Plds[r + 8 * hi][16 + lo] = (__bf16)p1;
    }
    __syncthreads();

    // ---- P (16x32) in A-matrix layout (contiguous k = 16h..16h+15 per lane)
    V16U pA;
    pA.q[0] = *(const v4u*)(&Plds[lo][kbB]);
    pA.q[1] = *(const v4u*)(&Plds[lo][kbB + 8]);

    // ---- P@V into acc (4 d-chunks); V transposed -> contiguous B loads
#pragma unroll
    for (int c2 = 0; c2 < 4; ++c2) {
      v16bf bv = *(const v16bf*)(vTB + (size_t)(c2 * 16 + lo) * TD + j0 + kbB);
      v8f& a = (c2 == 0) ? acc0 : (c2 == 1) ? acc1 : (c2 == 2) ? acc2 : acc3;
      a = __builtin_amdgcn_wmma_f32_16x16x32_bf16(false, pA.v, false, bv,
                                                  (short)0, a, false, false);
    }
    __syncthreads();
  }

  // ---- normalize and write context [S][B][H*DHD]
#pragma unroll
  for (int r = 0; r < 8; ++r) {
    float inv = 1.0f / lrow[r];
    int i = i0 + r + 8 * hi;
    float* orow = ctx + ((size_t)i * BATCH + b) * HD + h * DHD;
    orow[ 0 + lo] = acc0[r] * inv;
    orow[16 + lo] = acc1[r] * inv;
    orow[32 + lo] = acc2[r] * inv;
    orow[48 + lo] = acc3[r] * inv;
  }
}

// ---------------------------------------------------------------- launcher

extern "C" void kernel_launch(void* const* d_in, const int* in_sizes, int n_in,
                              void* d_out, int out_size, void* d_ws, size_t ws_size,
                              hipStream_t stream) {
  (void)in_sizes; (void)n_in; (void)out_size; (void)ws_size;
  const float* x_in = (const float*)d_in[0];   // [S][B][D]
  const float* pemb = (const float*)d_in[1];   // [T][D]
  const float* memp = (const float*)d_in[2];   // [M][B][D]
  const float* u_p  = (const float*)d_in[3];   // [H][DHD]
  const float* v_p  = (const float*)d_in[4];   // [H][DHD]
  const float* wkv  = (const float*)d_in[5];   // [D][2*HD]
  const float* wq   = (const float*)d_in[6];   // [D][HD]
  const float* wpos = (const float*)d_in[7];   // [D][HD]
  const float* wout = (const float*)d_in[8];   // [HD][D]
  float* out = (float*)d_out;                  // [S][B][D]

  char* ws = (char*)d_ws;
  size_t off = 0;
  auto alloc = [&](size_t bytes) -> void* {
    void* p = ws + off;
    off += (bytes + 255) & ~(size_t)255;
    return p;
  };

  __bf16* xcat_h  = (__bf16*)alloc((size_t)TD * BATCH * D_IN * 2);      // 8 MiB
  __bf16* wkvT_h  = (__bf16*)alloc((size_t)D_IN * N_KV * 2);            // 4 MiB
  __bf16* wqT_h   = (__bf16*)alloc((size_t)D_IN * HD * 2);              // 2 MiB
  __bf16* wposT_h = (__bf16*)alloc((size_t)D_IN * HD * 2);              // 2 MiB
  __bf16* woutT_h = (__bf16*)alloc((size_t)HD * D_IN * 2);              // 2 MiB
  __bf16* pemb_h  = (__bf16*)alloc((size_t)TD * D_IN * 2);              // 4 MiB
  float*  kv_f    = (float*) alloc((size_t)TD * BATCH * N_KV * 4);      // 32 MiB
  float*  q_f     = (float*) alloc((size_t)SEQ * BATCH * HD * 4);       // 8 MiB
  float*  r_f     = (float*) alloc((size_t)TD * HD * 4);                // 8 MiB
  __bf16* k_h     = (__bf16*)alloc((size_t)BATCH * NH * TD * DHD * 2);  // 8 MiB
  __bf16* vT_h    = (__bf16*)alloc((size_t)BATCH * NH * DHD * TD * 2);  // 8 MiB
  __bf16* qu_h    = (__bf16*)alloc((size_t)BATCH * NH * SEQ * DHD * 2); // 4 MiB
  float*  qv_f    = (float*) alloc((size_t)BATCH * NH * SEQ * DHD * 4); // 8 MiB
  float*  ctx_f   = (float*) alloc((size_t)SEQ * BATCH * HD * 4);       // 8 MiB
  __bf16* ctx_h   = (__bf16*)alloc((size_t)SEQ * BATCH * HD * 2);       // 4 MiB

  auto cvt = [&](const float* s, __bf16* d, int n) {
    xl_cvt_bf16<<<(n + 255) / 256, 256, 0, stream>>>(s, d, n);
  };
  auto cvtT = [&](const float* s, __bf16* d, int rows, int cols) {
    xl_cvtT_bf16<<<(rows * cols + 255) / 256, 256, 0, stream>>>(s, d, rows, cols);
  };

  // f32 -> bf16 staging (xcat = concat(memory, input) along rows)
  cvt(memp, xcat_h, MEMN * BATCH * D_IN);
  cvt(x_in, xcat_h + (size_t)MEMN * BATCH * D_IN, SEQ * BATCH * D_IN);
  cvt(pemb, pemb_h, TD * D_IN);
  // weights transposed for the WMMA B-operand layout
  cvtT(wkv,  wkvT_h,  D_IN, N_KV);
  cvtT(wq,   wqT_h,   D_IN, HD);
  cvtT(wpos, wposT_h, D_IN, HD);
  cvtT(wout, woutT_h, HD, D_IN);

  // projections (bf16 WMMA, f32 accumulate); grid = (N/64, M/16)
  xl_gemm_bf16<<<dim3(N_KV / 64, (TD * BATCH) / 16), 32, 0, stream>>>(
      xcat_h, wkvT_h, kv_f, TD * BATCH, N_KV, D_IN);
  xl_gemm_bf16<<<dim3(HD / 64, (SEQ * BATCH) / 16), 32, 0, stream>>>(
      xcat_h + (size_t)MEMN * BATCH * D_IN, wqT_h, q_f, SEQ * BATCH, HD, D_IN);
  xl_gemm_bf16<<<dim3(HD / 64, TD / 16), 32, 0, stream>>>(
      pemb_h, wposT_h, r_f, TD, HD, D_IN);

  // repack for attention
  xl_pack_kv<<<(TD * BATCH * HD + 255) / 256, 256, 0, stream>>>(kv_f, k_h, vT_h);
  xl_pack_q<<<(SEQ * BATCH * HD + 255) / 256, 256, 0, stream>>>(q_f, u_p, v_p, qu_h, qv_f);

  // fused attention
  xl_flash_attn<<<dim3(SEQ / 16, NH, BATCH), 32, 0, stream>>>(
      qu_h, qv_f, k_h, vT_h, r_f, ctx_f);

  // output projection
  cvt(ctx_f, ctx_h, SEQ * BATCH * HD);
  xl_gemm_bf16<<<dim3(D_IN / 64, (SEQ * BATCH) / 16), 32, 0, stream>>>(
      ctx_h, woutT_h, out, SEQ * BATCH, D_IN, HD);
}
